// SymplecticNeuralSheaf_2894807957872
// MI455X (gfx1250) — matrix operand order repaired and checked by reading, not compiled, not run on hardware
//
#include <hip/hip_runtime.h>
#include <hip/hip_bf16.h>

#define B_ROWS 16384
#define DIM    1024
#define DIM2   2048
#define HID    64
#define WAVES  4
#define TPB    (WAVES * 32)
#define ROWS_PER_WG (WAVES * 16)

typedef __attribute__((ext_vector_type(16))) _Float16 v16h;
typedef __attribute__((ext_vector_type(8)))  float    v8f;

// ---------------------------------------------------------------------------
// WMMA helper: D = A(16x32 f16) * B(32x16 f16) + C(16x16 f32)
// ---------------------------------------------------------------------------
__device__ __forceinline__ v8f wmma_f16(v16h a, v16h b, v8f c) {
    return __builtin_amdgcn_wmma_f32_16x16x32_f16(
        /*neg_a=*/false, a, /*neg_b=*/false, b,
        /*c_mod=*/(short)0, c, /*reuse_a=*/false, /*reuse_b=*/false);
}

__device__ __forceinline__ v8f zero8() {
    v8f z;
#pragma unroll
    for (int i = 0; i < 8; ++i) z[i] = 0.0f;
    return z;
}

// A-matrix (16x32 f16) build from a global f32 row.
// Lane l<16 holds row l: elems 0..15 = K k0..k0+15.
// Lane l>=16 holds row l-16 with an extra +16 on K (caller passes p += half*16).
__device__ __forceinline__ v16h make_a_f32(const float* p) {
    v16h a;
#pragma unroll
    for (int i = 0; i < 8; ++i) {
        a[i]     = (_Float16)p[i];
        a[8 + i] = (_Float16)p[8 + i];
    }
    return a;
}

// A-matrix pair (K=0..31 and K=32..63) from an LDS tile stored [16][64] f16.
__device__ __forceinline__ void make_a_lds(const _Float16* t, int lane16, int half,
                                           v16h& lo, v16h& hi) {
    const _Float16* r = t + lane16 * 64 + half * 16;
#pragma unroll
    for (int i = 0; i < 8; ++i) {
        lo[i]     = r[i];
        lo[8 + i] = r[8 + i];
        hi[i]     = r[32 + i];
        hi[8 + i] = r[40 + i];
    }
}

// One half (1024 wide) of a Z-pass: acc[0..3] += rows(src) x W1T[:, kOfs..]
__device__ __forceinline__ void z_half(const float* __restrict__ src,
                                       const _Float16* __restrict__ W1T,
                                       int kOfs, size_t rowOff,
                                       int lane16, int half, v8f acc[4]) {
#pragma unroll 2
    for (int k0 = 0; k0 < DIM; k0 += 32) {
        v16h a = make_a_f32(src + rowOff + k0 + half * 16);
#pragma unroll
        for (int nt = 0; nt < 4; ++nt) {
            // B 32x16: lane16 = column n, 16 contiguous f16 per lane (+16 K for hi half)
            v16h b = *(const v16h*)(W1T + (size_t)(nt * 16 + lane16) * DIM2
                                        + kOfs + k0 + half * 16);
            acc[nt] = wmma_f16(a, b, acc[nt]);
        }
    }
}

// K=64 matmul tile: [16x64] (A lo/hi) x Bbase[n][0..63] -> 16x16 f32
__device__ __forceinline__ v8f mm_k64(v16h alo, v16h ahi,
                                      const _Float16* __restrict__ Bbase,
                                      int n, int half) {
    v16h b0 = *(const v16h*)(Bbase + (size_t)n * 64 + half * 16);
    v16h b1 = *(const v16h*)(Bbase + (size_t)n * 64 + 32 + half * 16);
    v8f c = zero8();
    c = wmma_f16(alo, b0, c);
    c = wmma_f16(ahi, b1, c);
    return c;
}

// Elementwise pass on a 16x64 accumulator tile (C layout: lane16 = col, M = e + half*8).
// MODE 0: store h=gelu(z) and s=gelu'(z)*w2sum.  MODE 1: store s only.
// MODE 2: store d = h1(from LDS) - gelu(z).
template <int MODE>
__device__ __forceinline__ void activation_store(const v8f acc[4],
                                                 const float* __restrict__ b1,
                                                 const float* __restrict__ w2sum,
                                                 int lane16, int half,
                                                 _Float16* hT, _Float16* sT) {
#pragma unroll
    for (int nt = 0; nt < 4; ++nt) {
        const int n   = nt * 16 + lane16;
        const float bb = b1[n];
        const float ws = (MODE == 2) ? 0.0f : w2sum[n];
#pragma unroll
        for (int e = 0; e < 8; ++e) {
            const int M  = e + half * 8;
            const float z   = acc[nt][e] + bb;
            const float phi = 0.5f * (1.0f + erff(z * 0.70710678118654752f));
            if (MODE == 2) {
                const float h3 = z * phi;
                const float h1 = (float)hT[M * 64 + n];
                sT[M * 64 + n] = (_Float16)(h1 - h3);
            } else {
                const float pdf = 0.3989422804014327f * __expf(-0.5f * z * z);
                sT[M * 64 + n] = (_Float16)((phi + z * pdf) * ws);
                if (MODE == 0) hT[M * 64 + n] = (_Float16)(z * phi);
            }
        }
    }
}

// ---------------------------------------------------------------------------
// Prologue kernels: convert weights to f16 in WMMA-B-friendly layouts.
// ---------------------------------------------------------------------------
__global__ void prep_w1t(const float* __restrict__ W1, _Float16* __restrict__ W1T) {
    int idx = blockIdx.x * 256 + threadIdx.x;          // 64*2048
    if (idx < HID * DIM2) {
        int n = idx / DIM2, k = idx % DIM2;
        W1T[idx] = (_Float16)W1[(size_t)k * HID + n];  // W1T[n][k] = W1[k][n]
    }
}
__global__ void prep_w1q(const float* __restrict__ W1, _Float16* __restrict__ W1q) {
    int idx = blockIdx.x * 256 + threadIdx.x;          // 1024*64 (first DIM rows, row-major)
    if (idx < DIM * HID) W1q[idx] = (_Float16)W1[idx];
}
__global__ void prep_w2t(const float* __restrict__ W2, _Float16* __restrict__ W2T) {
    int idx = blockIdx.x * 256 + threadIdx.x;          // 1024*64
    if (idx < DIM * HID) {
        int n = idx >> 6, k = idx & 63;
        W2T[idx] = (_Float16)W2[(size_t)k * DIM + n];  // W2T[n][k] = W2[k][n]
    }
}
__global__ void prep_w2sum(const float* __restrict__ W2, float* __restrict__ w2sum) {
    int j = threadIdx.x;                               // 64 threads
    if (j < HID) {
        float s = 0.0f;
        for (int k = 0; k < DIM; ++k) s += W2[(size_t)j * DIM + k];
        w2sum[j] = s;
    }
}

// ---------------------------------------------------------------------------
// Main fused leapfrog kernel: one wave owns 16 rows end-to-end.
// out_e is used as temporary storage for p_half, overwritten by energy at end.
// ---------------------------------------------------------------------------
__global__ __launch_bounds__(TPB) void hnn_leapfrog(
    const float* __restrict__ q, const float* __restrict__ p,
    const float* __restrict__ b1, const float* __restrict__ mass,
    const float* __restrict__ dt_param,
    const _Float16* __restrict__ W1T, const _Float16* __restrict__ W1q,
    const _Float16* __restrict__ W2T, const float* __restrict__ w2sum,
    float* __restrict__ out_q, float* __restrict__ out_p, float* __restrict__ out_e)
{
    __shared__ _Float16 sh_h[WAVES][16 * 64];
    __shared__ _Float16 sh_s[WAVES][16 * 64];

    const int lane   = threadIdx.x & 31;
    const int wv     = threadIdx.x >> 5;
    const int lane16 = lane & 15;
    const int half   = lane >> 4;
    const int rbase  = (blockIdx.x * WAVES + wv) * 16;
    const size_t rowOff = (size_t)(rbase + lane16) * DIM;   // A-source row for this lane

    const float dt = 1.0f / (1.0f + __expf(-dt_param[0]));  // sigmoid(dt_param)
    const float c1 = 0.5f * dt;

    _Float16* hT = sh_h[wv];
    _Float16* sT = sh_s[wv];

    // ---- Phase 1: z1 = [q,p] @ W1 ; h1,s1 -> LDS ------------------------
    v8f acc[4] = {zero8(), zero8(), zero8(), zero8()};
    z_half(q, W1T, 0,   rowOff, lane16, half, acc);
    z_half(p, W1T, DIM, rowOff, lane16, half, acc);
    activation_store<0>(acc, b1, w2sum, lane16, half, hT, sT);
    v16h As_lo, As_hi;
    make_a_lds(sT, lane16, half, As_lo, As_hi);

    // ---- Phase 2: dH_dq = s1 @ W1q^T ; p_half, q_new --------------------
    for (int nt = 0; nt < DIM / 16; ++nt) {
        const int n = nt * 16 + lane16;
        v8f c = mm_k64(As_lo, As_hi, W1q, n, half);
        const float im = dt / mass[n];
#pragma unroll
        for (int e = 0; e < 8; ++e) {
            const size_t addr = (size_t)(rbase + e + half * 8) * DIM + n;
            const float ph = p[addr] - c1 * c[e];
            out_q[addr] = q[addr] + im * ph;
            out_e[addr] = ph;                    // stage p_half
        }
    }

    // ---- Phase 3: z2 = [q_new, p_half] @ W1 ; s2 -> LDS -----------------
    v8f acc2[4] = {zero8(), zero8(), zero8(), zero8()};
    z_half(out_q, W1T, 0,   rowOff, lane16, half, acc2);
    z_half(out_e, W1T, DIM, rowOff, lane16, half, acc2);
    activation_store<1>(acc2, b1, w2sum, lane16, half, hT, sT);
    v16h As2_lo, As2_hi;
    make_a_lds(sT, lane16, half, As2_lo, As2_hi);

    // ---- Phase 4: dH_dq2 ; p_new ----------------------------------------
    for (int nt = 0; nt < DIM / 16; ++nt) {
        const int n = nt * 16 + lane16;
        v8f c = mm_k64(As2_lo, As2_hi, W1q, n, half);
#pragma unroll
        for (int e = 0; e < 8; ++e) {
            const size_t addr = (size_t)(rbase + e + half * 8) * DIM + n;
            out_p[addr] = out_e[addr] - c1 * c[e];
        }
    }

    // ---- Phase 5: z3 = [q_new, p_new] @ W1 ; d = h1 - h3 -> LDS ---------
    v8f acc3[4] = {zero8(), zero8(), zero8(), zero8()};
    z_half(out_q, W1T, 0,   rowOff, lane16, half, acc3);
    z_half(out_p, W1T, DIM, rowOff, lane16, half, acc3);
    activation_store<2>(acc3, b1, w2sum, lane16, half, hT, sT);
    v16h Ad_lo, Ad_hi;
    make_a_lds(sT, lane16, half, Ad_lo, Ad_hi);

    // ---- Phase 6: energy = |(h1 - h3) @ W2| (b2 cancels) ----------------
    for (int nt = 0; nt < DIM / 16; ++nt) {
        const int n = nt * 16 + lane16;
        v8f c = mm_k64(Ad_lo, Ad_hi, W2T, n, half);
#pragma unroll
        for (int e = 0; e < 8; ++e) {
            const size_t addr = (size_t)(rbase + e + half * 8) * DIM + n;
            out_e[addr] = fabsf(c[e]);
        }
    }
}

// ---------------------------------------------------------------------------
extern "C" void kernel_launch(void* const* d_in, const int* in_sizes, int n_in,
                              void* d_out, int out_size, void* d_ws, size_t ws_size,
                              hipStream_t stream) {
    (void)in_sizes; (void)n_in; (void)out_size; (void)ws_size;

    const float* q    = (const float*)d_in[0];
    const float* p    = (const float*)d_in[1];
    const float* W1   = (const float*)d_in[2];
    const float* b1   = (const float*)d_in[3];
    const float* W2   = (const float*)d_in[4];
    // d_in[5] = b2 (cancels analytically)
    const float* mass = (const float*)d_in[6];
    const float* dtp  = (const float*)d_in[7];

    char* ws = (char*)d_ws;
    _Float16* W1T  = (_Float16*)(ws);                                   // 64*2048 f16
    _Float16* W1q  = (_Float16*)(ws + (size_t)HID * DIM2 * 2);          // 1024*64 f16
    _Float16* W2T  = (_Float16*)(ws + (size_t)HID * DIM2 * 2 + (size_t)DIM * HID * 2);
    float*    w2sm = (float*)   (ws + (size_t)HID * DIM2 * 2 + (size_t)2 * DIM * HID * 2);

    float* out_q = (float*)d_out;
    float* out_p = out_q + (size_t)B_ROWS * DIM;
    float* out_e = out_p + (size_t)B_ROWS * DIM;

    prep_w1t<<<(HID * DIM2 + 255) / 256, 256, 0, stream>>>(W1, W1T);
    prep_w1q<<<(DIM * HID + 255) / 256, 256, 0, stream>>>(W1, W1q);
    prep_w2t<<<(DIM * HID + 255) / 256, 256, 0, stream>>>(W2, W2T);
    prep_w2sum<<<1, 64, 0, stream>>>(W2, w2sm);

    hnn_leapfrog<<<B_ROWS / ROWS_PER_WG, TPB, 0, stream>>>(
        q, p, b1, mass, dtp, W1T, W1q, W2T, w2sm, out_q, out_p, out_e);
}
